// MultiHeadSelfAttention_11639361372176
// MI455X (gfx1250) — matrix operand order repaired and checked by reading, not compiled, not run on hardware
//
#include <hip/hip_runtime.h>
#include <hip/hip_bf16.h>
#include <stdint.h>

// ---------------------------------------------------------------------------
// Types
// ---------------------------------------------------------------------------
typedef __bf16 bf16_t;
typedef __attribute__((ext_vector_type(16))) __bf16 bf16x16;
typedef __attribute__((ext_vector_type(8)))  __bf16 bf16x8;
typedef __attribute__((ext_vector_type(8)))  float  f32x8;
typedef __attribute__((ext_vector_type(4)))  unsigned int u32x4;
typedef __attribute__((ext_vector_type(8)))  int    i32x8;
typedef __attribute__((ext_vector_type(4)))  int    i32x4;

#define B_SZ     8
#define N_SEQ    1024
#define D_MODEL  768
#define N_HEADS  12
#define HEAD_DIM 64
#define SCALE    0.125f            // 64^-0.5
#define QKV_N    (3 * D_MODEL)     // 2304
#define M_ROWS   (B_SZ * N_SEQ)    // 8192

// ---------------------------------------------------------------------------
// Helpers
// ---------------------------------------------------------------------------
__device__ __forceinline__ bf16_t f2bf(float f) {
  union { float f; uint32_t u; } a; a.f = f;
  uint32_t r = (a.u + 0x7FFFu + ((a.u >> 16) & 1u)) >> 16;
  union { uint16_t u; bf16_t b; } o; o.u = (uint16_t)r;
  return o.b;
}

union FragU { bf16x16 v; bf16x8 h[2]; };

// A-operand fragment (16x32 bf16) from row-major [16][>=32] tile, row stride ldk.
__device__ __forceinline__ bf16x16 load_frag_a(const bf16_t* base, int ldk, int lane) {
  const int half = lane >> 4, r = lane & 15;
  const bf16_t* p = base + (size_t)r * ldk + half * 8;
  FragU f;
  f.h[0] = *(const bf16x8*)(p);
  f.h[1] = *(const bf16x8*)(p + 16);
  return f.v;
}

// B-operand fragment (32x16 bf16) loaded from B^T stored row-major [16][>=32].
__device__ __forceinline__ bf16x16 load_frag_b(const bf16_t* base, int ldk, int lane) {
  const int half = lane >> 4, r = lane & 15;
  const bf16_t* p = base + (size_t)r * ldk + half * 8;
  FragU f;
  f.h[0] = *(const bf16x8*)(p);
  f.h[1] = *(const bf16x8*)(p + 16);
  return f.v;
}

__device__ __forceinline__ f32x8 wmma_bf16(bf16x16 a, bf16x16 b, f32x8 c) {
  return __builtin_amdgcn_wmma_f32_16x16x32_bf16(false, a, false, b, (short)0, c,
                                                 false, false);
}

__device__ __forceinline__ void wait_ds0() {
  asm volatile("s_wait_dscnt 0x0" ::: "memory");
}

// Low 32 bits of a generic LDS pointer = workgroup-relative LDS byte address
// (flat-aperture truncation rule, ISA 10.2).
__device__ __forceinline__ uint32_t lds_byte_addr(const void* p) {
  return (uint32_t)(uintptr_t)p;
}

// ---------------------------------------------------------------------------
// TDM: issue tensor_load_to_lds for one 128x32 bf16 tile (row-major, row
// stride = strideElems). 2-D descriptor: groups 2/3 zero.  (ISA 08, sec 8)
// ---------------------------------------------------------------------------
__device__ __forceinline__ void tdm_load_tile_128x32(uint32_t lds_addr,
                                                     const bf16_t* gptr,
                                                     uint32_t strideElems) {
  uint64_t ga = (uint64_t)(uintptr_t)gptr;
  u32x4 g0;
  g0[0] = 1u;                                        // count=1, user descriptor
  g0[1] = lds_addr;                                  // lds_addr (bytes)
  g0[2] = (uint32_t)ga;                              // global_addr[31:0]
  g0[3] = (uint32_t)((ga >> 32) & 0x01FFFFFFu)       // global_addr[56:32]
        | (2u << 30);                                // type = 2 ("image")
  i32x8 g1;
  g1[0] = (int)(1u << 16);                           // data_size=1 -> 2 bytes
  g1[1] = (int)(32u << 16);                          // tensor_dim0 = 32
  g1[2] = (int)(128u << 16);                         // tensor_dim1 = 128
  g1[3] = (int)(32u << 16);                          // tile_dim0 = 32 (dim1 hi = 0)
  g1[4] = (int)128u;                                 // tile_dim1 = 128, tile_dim2 = 0
  g1[5] = (int)strideElems;                          // tensor_dim0_stride lo
  g1[6] = 0;                                         // stride hi / dim1_stride lo
  g1[7] = 0;
  i32x4 z4 = (i32x4){0, 0, 0, 0};
#if defined(__clang_major__) && (__clang_major__ >= 23)
  i32x8 z8 = (i32x8){0, 0, 0, 0, 0, 0, 0, 0};
  __builtin_amdgcn_tensor_load_to_lds(g0, g1, z4, z4, z8, 0);
#else
  __builtin_amdgcn_tensor_load_to_lds(g0, g1, z4, z4, 0);
#endif
}

// ---------------------------------------------------------------------------
// Kernel 1: fp32 -> bf16 straight copy (x)
// ---------------------------------------------------------------------------
__global__ void f2bf_copy_kernel(const float* __restrict__ in,
                                 bf16_t* __restrict__ out, int n) {
  int i = blockIdx.x * 256 + threadIdx.x;
  if (i < n) out[i] = f2bf(in[i]);
}

// ---------------------------------------------------------------------------
// Kernel 2: fp32 [R][C] -> bf16 transposed [C][R]   (weights)
// ---------------------------------------------------------------------------
__global__ void f2bf_transpose_kernel(const float* __restrict__ in,
                                      bf16_t* __restrict__ out, int R, int C) {
  __shared__ float tile[32][33];
  const int bx = blockIdx.x * 32;
  const int by = blockIdx.y * 32;
  const int tx = threadIdx.x & 31;
  const int ty = threadIdx.x >> 5;
#pragma unroll
  for (int i = 0; i < 32; i += 8) {
    int r = by + ty + i, c = bx + tx;
    tile[ty + i][tx] = (r < R && c < C) ? in[(size_t)r * C + c] : 0.f;
  }
  __syncthreads();
#pragma unroll
  for (int i = 0; i < 32; i += 8) {
    int c = bx + ty + i, r = by + tx;
    if (c < C && r < R) out[(size_t)c * R + r] = f2bf(tile[tx][ty + i]);
  }
}

// ---------------------------------------------------------------------------
// Kernel 3: QKV GEMM (M=8192,N=2304,K=768), TDM-staged LDS tiles, double
// buffered; epilogue scatters Q/K bf16 [B,H,N,64] and V transposed [B,H,64,N].
// ---------------------------------------------------------------------------
__global__ __launch_bounds__(256)
void qkv_gemm_kernel(const bf16_t* __restrict__ A, const bf16_t* __restrict__ BT,
                     bf16_t* __restrict__ Qb, bf16_t* __restrict__ Kb,
                     bf16_t* __restrict__ Vt) {
  __shared__ __align__(16) bf16_t As[2][128 * 32];
  __shared__ __align__(16) bf16_t Bs[2][128 * 32];
  const int tid = threadIdx.x, lane = tid & 31, wave = tid >> 5;
  const int wm = wave >> 2, wn = wave & 3;
  const int m0 = blockIdx.y * 128;
  const int n0 = blockIdx.x * 128;
  const int NK = D_MODEL / 32;                       // 24 k-steps

  f32x8 acc[4][2];
#pragma unroll
  for (int i = 0; i < 4; ++i)
#pragma unroll
    for (int j = 0; j < 2; ++j) acc[i][j] = (f32x8){0,0,0,0,0,0,0,0};

  if (wave == 0) {
    tdm_load_tile_128x32(lds_byte_addr(&As[0][0]), &A[(size_t)m0 * D_MODEL], D_MODEL);
    tdm_load_tile_128x32(lds_byte_addr(&Bs[0][0]), &BT[(size_t)n0 * D_MODEL], D_MODEL);
  }

  for (int it = 0; it < NK; ++it) {
    if (wave == 0) {
      if (it + 1 < NK) {                 // prefetch next tile into other buffer
        int k1 = (it + 1) * 32;
        tdm_load_tile_128x32(lds_byte_addr(&As[(it + 1) & 1][0]),
                             &A[(size_t)m0 * D_MODEL + k1], D_MODEL);
        tdm_load_tile_128x32(lds_byte_addr(&Bs[(it + 1) & 1][0]),
                             &BT[(size_t)n0 * D_MODEL + k1], D_MODEL);
        __builtin_amdgcn_s_wait_tensorcnt(2);   // current tile's 2 DMAs done
      } else {
        __builtin_amdgcn_s_wait_tensorcnt(0);
      }
    }
    __syncthreads();

    const bf16_t* Asb = As[it & 1];
    const bf16_t* Bsb = Bs[it & 1];
    bf16x16 af[4], bfr[2];
#pragma unroll
    for (int i = 0; i < 4; ++i)
      af[i] = load_frag_a(&Asb[(wm * 64 + i * 16) * 32], 32, lane);
#pragma unroll
    for (int j = 0; j < 2; ++j)
      bfr[j] = load_frag_b(&Bsb[(wn * 32 + j * 16) * 32], 32, lane);
#pragma unroll
    for (int i = 0; i < 4; ++i)
#pragma unroll
      for (int j = 0; j < 2; ++j) acc[i][j] = wmma_bf16(af[i], bfr[j], acc[i][j]);
    __syncthreads();
  }

  // Epilogue: scatter into Q / K / V(transposed) as bf16.
  const int half = lane >> 4, nl = lane & 15;
#pragma unroll
  for (int i = 0; i < 4; ++i)
#pragma unroll
    for (int j = 0; j < 2; ++j) {
      int n = n0 + wn * 32 + j * 16 + nl;
      int which = n / D_MODEL;                // 0=q 1=k 2=v
      int d = n % D_MODEL, h = d >> 6, hd = d & 63;
#pragma unroll
      for (int r = 0; r < 8; ++r) {
        int m = m0 + wm * 64 + i * 16 + r + half * 8;
        int b = m >> 10, row = m & 1023;
        bf16_t v = f2bf(acc[i][j][r]);
        size_t bh = (size_t)(b * N_HEADS + h);
        if (which == 0)      Qb[(bh * N_SEQ + row) * HEAD_DIM + hd] = v;
        else if (which == 1) Kb[(bh * N_SEQ + row) * HEAD_DIM + hd] = v;
        else                 Vt[(bh * HEAD_DIM + hd) * N_SEQ + row] = v;
      }
    }
}

// ---------------------------------------------------------------------------
// Kernel 4: flash attention. grid = (N/128, B*H), 256 threads = 8 waves,
// one wave owns 16 query rows; online softmax over 32-key blocks.
// ---------------------------------------------------------------------------
__global__ __launch_bounds__(256)
void attn_kernel(const bf16_t* __restrict__ Qb, const bf16_t* __restrict__ Kb,
                 const bf16_t* __restrict__ Vt, bf16_t* __restrict__ AO) {
  __shared__ __align__(16) float  Sbuf[8][16][32];
  __shared__ __align__(16) bf16_t Pbuf[8][16][32];
  __shared__ float sm[8][16], sl[8][16], ssc[8][16];

  const int tid = threadIdx.x, lane = tid & 31, wave = tid >> 5;
  const int half = lane >> 4, nl = lane & 15;
  const int bh = blockIdx.y;
  const int b = bh / N_HEADS, h = bh % N_HEADS;
  const int q0 = blockIdx.x * 128 + wave * 16;

  const bf16_t* Qp = Qb + (size_t)bh * N_SEQ * HEAD_DIM;
  const bf16_t* Kp = Kb + (size_t)bh * N_SEQ * HEAD_DIM;
  const bf16_t* Vp = Vt + (size_t)bh * HEAD_DIM * N_SEQ;

  bf16x16 qa0 = load_frag_a(Qp + (size_t)q0 * HEAD_DIM + 0,  HEAD_DIM, lane);
  bf16x16 qa1 = load_frag_a(Qp + (size_t)q0 * HEAD_DIM + 32, HEAD_DIM, lane);

  f32x8 o[4];
#pragma unroll
  for (int t = 0; t < 4; ++t) o[t] = (f32x8){0,0,0,0,0,0,0,0};
  if (lane < 16) { sm[wave][lane] = -1e30f; sl[wave][lane] = 0.f; }
  wait_ds0();

  for (int kb = 0; kb < N_SEQ; kb += 32) {
    // ---- S = (Q K^T) * scale for 16 rows x 32 keys --------------------
    bf16x16 kf00 = load_frag_b(Kp + (size_t)(kb +  0) * HEAD_DIM +  0, HEAD_DIM, lane);
    bf16x16 kf01 = load_frag_b(Kp + (size_t)(kb +  0) * HEAD_DIM + 32, HEAD_DIM, lane);
    bf16x16 kf10 = load_frag_b(Kp + (size_t)(kb + 16) * HEAD_DIM +  0, HEAD_DIM, lane);
    bf16x16 kf11 = load_frag_b(Kp + (size_t)(kb + 16) * HEAD_DIM + 32, HEAD_DIM, lane);
    f32x8 s0 = (f32x8){0,0,0,0,0,0,0,0};
    f32x8 s1 = (f32x8){0,0,0,0,0,0,0,0};
    s0 = wmma_bf16(qa0, kf00, s0);  s0 = wmma_bf16(qa1, kf01, s0);
    s1 = wmma_bf16(qa0, kf10, s1);  s1 = wmma_bf16(qa1, kf11, s1);

#pragma unroll
    for (int r = 0; r < 8; ++r) {
      Sbuf[wave][r + half * 8][nl]      = s0[r] * SCALE;
      Sbuf[wave][r + half * 8][16 + nl] = s1[r] * SCALE;
    }
    wait_ds0();

    // ---- online softmax over this 32-key block (one lane per row) -----
    if (lane < 16) {
      float mo = sm[wave][lane];
      float sv[32], bmax = -1e30f;
#pragma unroll
      for (int c = 0; c < 32; ++c) { sv[c] = Sbuf[wave][lane][c]; bmax = fmaxf(bmax, sv[c]); }
      float nm = fmaxf(mo, bmax);
      float rs = 0.f;
#pragma unroll
      for (int c = 0; c < 32; ++c) {
        float p = __expf(sv[c] - nm);
        rs += p;
        Pbuf[wave][lane][c] = f2bf(p);
      }
      float sc = __expf(mo - nm);
      sm[wave][lane] = nm;
      sl[wave][lane] = sl[wave][lane] * sc + rs;
      ssc[wave][lane] = sc;
    }
    wait_ds0();

    // ---- rescale O and accumulate P * V -------------------------------
    float fr[8];
#pragma unroll
    for (int r = 0; r < 8; ++r) fr[r] = ssc[wave][r + half * 8];
#pragma unroll
    for (int t = 0; t < 4; ++t)
#pragma unroll
      for (int r = 0; r < 8; ++r) o[t][r] *= fr[r];

    bf16x16 pa = load_frag_a(&Pbuf[wave][0][0], 32, lane);
#pragma unroll
    for (int t = 0; t < 4; ++t) {
      bf16x16 vf = load_frag_b(Vp + (size_t)(t * 16) * N_SEQ + kb, N_SEQ, lane);
      o[t] = wmma_bf16(pa, vf, o[t]);
    }
  }

  wait_ds0();
  float linv[8];
#pragma unroll
  for (int r = 0; r < 8; ++r) linv[r] = 1.f / sl[wave][r + half * 8];
#pragma unroll
  for (int t = 0; t < 4; ++t)
#pragma unroll
    for (int r = 0; r < 8; ++r) {
      int m = q0 + r + half * 8;
      AO[((size_t)b * N_SEQ + m) * D_MODEL + h * HEAD_DIM + t * 16 + nl] =
          f2bf(o[t][r] * linv[r]);
    }
}

// ---------------------------------------------------------------------------
// Kernel 5: output projection GEMM (M=8192,N=768,K=768) + bias, TDM staged.
// ---------------------------------------------------------------------------
__global__ __launch_bounds__(256)
void proj_gemm_kernel(const bf16_t* __restrict__ A, const bf16_t* __restrict__ BT,
                      const float* __restrict__ bias, float* __restrict__ Out) {
  __shared__ __align__(16) bf16_t As[2][128 * 32];
  __shared__ __align__(16) bf16_t Bs[2][128 * 32];
  const int tid = threadIdx.x, lane = tid & 31, wave = tid >> 5;
  const int wm = wave >> 2, wn = wave & 3;
  const int m0 = blockIdx.y * 128;
  const int n0 = blockIdx.x * 128;
  const int NK = D_MODEL / 32;

  f32x8 acc[4][2];
#pragma unroll
  for (int i = 0; i < 4; ++i)
#pragma unroll
    for (int j = 0; j < 2; ++j) acc[i][j] = (f32x8){0,0,0,0,0,0,0,0};

  if (wave == 0) {
    tdm_load_tile_128x32(lds_byte_addr(&As[0][0]), &A[(size_t)m0 * D_MODEL], D_MODEL);
    tdm_load_tile_128x32(lds_byte_addr(&Bs[0][0]), &BT[(size_t)n0 * D_MODEL], D_MODEL);
  }

  for (int it = 0; it < NK; ++it) {
    if (wave == 0) {
      if (it + 1 < NK) {
        int k1 = (it + 1) * 32;
        tdm_load_tile_128x32(lds_byte_addr(&As[(it + 1) & 1][0]),
                             &A[(size_t)m0 * D_MODEL + k1], D_MODEL);
        tdm_load_tile_128x32(lds_byte_addr(&Bs[(it + 1) & 1][0]),
                             &BT[(size_t)n0 * D_MODEL + k1], D_MODEL);
        __builtin_amdgcn_s_wait_tensorcnt(2);
      } else {
        __builtin_amdgcn_s_wait_tensorcnt(0);
      }
    }
    __syncthreads();

    const bf16_t* Asb = As[it & 1];
    const bf16_t* Bsb = Bs[it & 1];
    bf16x16 af[4], bfr[2];
#pragma unroll
    for (int i = 0; i < 4; ++i)
      af[i] = load_frag_a(&Asb[(wm * 64 + i * 16) * 32], 32, lane);
#pragma unroll
    for (int j = 0; j < 2; ++j)
      bfr[j] = load_frag_b(&Bsb[(wn * 32 + j * 16) * 32], 32, lane);
#pragma unroll
    for (int i = 0; i < 4; ++i)
#pragma unroll
      for (int j = 0; j < 2; ++j) acc[i][j] = wmma_bf16(af[i], bfr[j], acc[i][j]);
    __syncthreads();
  }

  const int half = lane >> 4, nl = lane & 15;
#pragma unroll
  for (int i = 0; i < 4; ++i)
#pragma unroll
    for (int j = 0; j < 2; ++j) {
      int n = n0 + wn * 32 + j * 16 + nl;
      float bn = bias[n];
#pragma unroll
      for (int r = 0; r < 8; ++r) {
        int m = m0 + wm * 64 + i * 16 + r + half * 8;
        Out[(size_t)m * D_MODEL + n] = acc[i][j][r] + bn;
      }
    }
}

// ---------------------------------------------------------------------------
// Launch
// ---------------------------------------------------------------------------
extern "C" void kernel_launch(void* const* d_in, const int* in_sizes, int n_in,
                              void* d_out, int out_size, void* d_ws, size_t ws_size,
                              hipStream_t stream) {
  (void)in_sizes; (void)n_in; (void)out_size; (void)ws_size;
  const float* x     = (const float*)d_in[0];
  const float* Wqkv  = (const float*)d_in[1];
  const float* Wproj = (const float*)d_in[2];
  const float* bproj = (const float*)d_in[3];
  float* out = (float*)d_out;

  bf16_t* xb     = (bf16_t*)d_ws;                          // 8192*768
  bf16_t* WqkvT  = xb     + (size_t)M_ROWS * D_MODEL;      // 2304*768
  bf16_t* WprojT = WqkvT  + (size_t)QKV_N * D_MODEL;       // 768*768
  bf16_t* Qb     = WprojT + (size_t)D_MODEL * D_MODEL;     // 96*1024*64
  bf16_t* Kb     = Qb     + (size_t)B_SZ * N_HEADS * N_SEQ * HEAD_DIM;
  bf16_t* Vt     = Kb     + (size_t)B_SZ * N_HEADS * N_SEQ * HEAD_DIM;
  bf16_t* AO     = Vt     + (size_t)B_SZ * N_HEADS * N_SEQ * HEAD_DIM;

  {
    int n = M_ROWS * D_MODEL;
    f2bf_copy_kernel<<<(n + 255) / 256, 256, 0, stream>>>(x, xb, n);
  }
  f2bf_transpose_kernel<<<dim3(QKV_N / 32, D_MODEL / 32), 256, 0, stream>>>(
      Wqkv, WqkvT, D_MODEL, QKV_N);
  f2bf_transpose_kernel<<<dim3(D_MODEL / 32, D_MODEL / 32), 256, 0, stream>>>(
      Wproj, WprojT, D_MODEL, D_MODEL);

  qkv_gemm_kernel<<<dim3(QKV_N / 128, M_ROWS / 128), 256, 0, stream>>>(
      xb, WqkvT, Qb, Kb, Vt);

  attn_kernel<<<dim3(N_SEQ / 128, B_SZ * N_HEADS), 256, 0, stream>>>(Qb, Kb, Vt, AO);

  proj_gemm_kernel<<<dim3(D_MODEL / 128, M_ROWS / 128), 256, 0, stream>>>(
      AO, WprojT, bproj, out);
}
